// PCRGLite_11252814315864
// MI455X (gfx1250) — matrix operand order repaired and checked by reading, not compiled, not run on hardware
//
#include <hip/hip_runtime.h>
#include <hip/hip_bf16.h>

typedef __bf16 bf16_t;
typedef __attribute__((ext_vector_type(16))) __bf16 v16bf;
typedef __attribute__((ext_vector_type(8)))  __bf16 v8bf;
typedef __attribute__((ext_vector_type(8)))  float  v8f;

#define NB   512
#define NL   256
#define ND   128
#define NG   4
#define NH1  64
#define NH2  32

// ---------------------------------------------------------------------------
// K0: fold/convert weights to bf16 in workspace.
//   Wm  [64][128] = sm_w1[:, 384:512]          (q*k block)
//   Wkd [64][128] = sm_w1[:,128:256] - [:,256:384]   (k - d block)
//   W2b [32][64]  = sm_w2
// Row-major so a WMMA B-fragment (16 contiguous K for fixed n) is one v16bf.
// ---------------------------------------------------------------------------
__global__ void prep_weights(const float* __restrict__ sm_w1,
                             const float* __restrict__ sm_w2,
                             bf16_t* __restrict__ Wm,
                             bf16_t* __restrict__ Wkd,
                             bf16_t* __restrict__ W2b) {
  int tid = threadIdx.x;
  for (int idx = tid; idx < NH1 * ND; idx += blockDim.x) {
    int n = idx >> 7, k = idx & 127;
    const float* row = sm_w1 + (size_t)n * 4 * ND;
    Wm[idx]  = (bf16_t)row[3 * ND + k];
    Wkd[idx] = (bf16_t)(row[ND + k] - row[2 * ND + k]);
  }
  for (int idx = tid; idx < NH2 * NH1; idx += blockDim.x)
    W2b[idx] = (bf16_t)sm_w2[idx];
}

// ---------------------------------------------------------------------------
// K1: per-batch query generation MLP (fp32, trivial FLOPs) +
//     per-(b,g) folded bias t[g][n] = (Wq+Wd)[n,:] . q_item[g] + b1[n]
// ---------------------------------------------------------------------------
__global__ void gen_queries(const float* __restrict__ cand, const float* __restrict__ ctx,
                            const float* __restrict__ w1, const float* __restrict__ b1,
                            const float* __restrict__ w2, const float* __restrict__ b2,
                            const float* __restrict__ w3, const float* __restrict__ b3,
                            const float* __restrict__ q2w, const float* __restrict__ q2b,
                            const float* __restrict__ smw1, const float* __restrict__ smb1,
                            float* __restrict__ q_item_g, float* __restrict__ t_g) {
  __shared__ float qin[256], h1[64], h2[32], qv[192], qitem[512];
  int b = blockIdx.x, tid = threadIdx.x;
  for (int i = tid; i < 256; i += 64)
    qin[i] = (i < 128) ? cand[b * 128 + i] : ctx[b * 128 + i - 128];
  __syncthreads();
  {
    float s = b1[tid];
    for (int j = 0; j < 256; ++j) s += w1[tid * 256 + j] * qin[j];
    h1[tid] = fmaxf(s, 0.f);
  }
  __syncthreads();
  if (tid < 32) {
    float s = b2[tid];
    for (int j = 0; j < 64; ++j) s += w2[tid * 64 + j] * h1[j];
    h2[tid] = fmaxf(s, 0.f);
  }
  __syncthreads();
  for (int i = tid; i < 192; i += 64) {
    float s = b3[i];
    for (int j = 0; j < 32; ++j) s += w3[i * 32 + j] * h2[j];
    qv[i] = s;
  }
  __syncthreads();
  for (int i = tid; i < 512; i += 64) {
    int g = i >> 7, d = i & 127;
    float s = q2b[d];
    for (int j = 0; j < 48; ++j) s += q2w[d * 48 + j] * qv[g * 48 + j];
    qitem[i] = s;
    q_item_g[(size_t)b * 512 + i] = s;
  }
  __syncthreads();
  for (int i = tid; i < 256; i += 64) {
    int g = i >> 6, n = i & 63;
    float s = smb1[n];
    const float* r = smw1 + (size_t)n * 512;
    for (int d = 0; d < 128; ++d) s += (r[d] + r[256 + d]) * qitem[g * 128 + d];
    t_g[(size_t)b * 256 + i] = s;
  }
}

// ---------------------------------------------------------------------------
// K2: main kernel. One block (8 wave32) per batch element.
//   phase1: u[l][n] = Wkd @ hist[l]        (WMMA, [256,128]x[128,64])
//   phase2: h1 = relu(t[g] + u[l] + Wm @ (q[g]*hist[l]))  (WMMA, [1024,128]x[128,64])
//           h2 = h1 @ W2^T + b2           (WMMA, K=64, N=32)
//           score = h2 . w3 + b3           (shfl_xor half-wave reduction)
//   phase3: masked softmax over L per (g)
//   phase4: z[g] = probs @ hist (fp32 from global), mean over g -> out
// ---------------------------------------------------------------------------
__launch_bounds__(256)
__global__ void din_main(const float* __restrict__ hist, const int* __restrict__ mask,
                         const float* __restrict__ q_item_g, const float* __restrict__ t_g,
                         const bf16_t* __restrict__ Wm, const bf16_t* __restrict__ Wkd,
                         const bf16_t* __restrict__ W2b,
                         const float* __restrict__ sm_b2, const float* __restrict__ sm_w3,
                         const float* __restrict__ sm_b3,
                         float* __restrict__ out) {
  extern __shared__ char smem[];
  bf16_t* histb = (bf16_t*)smem;              // [256][128] bf16 : 65536 B
  float*  u     = (float*)(smem + 65536);     // [256][64]  f32  : 65536 B
  float*  qi    = (float*)(smem + 131072);    // [4][128]   f32  : 2048  B
  float*  tt    = (float*)(smem + 133120);    // [4][64]    f32  : 1024  B
  float*  sc    = (float*)(smem + 134144);    // [4][256]   f32  : 4096  B
  bf16_t* h1s   = (bf16_t*)(smem + 138240);   // [8][16][64] bf16: 16384 B (per-wave staging)
  float*  zb    = (float*)(smem + 154624);    // [4][128]   f32  : 2048  B

  const int b = blockIdx.x, tid = threadIdx.x;
  const int wave = tid >> 5, lane = tid & 31;
  const int lm = lane & 15, khalf = lane >> 4;

  // ---- load hist[b] -> LDS as bf16; q_item, t -> LDS fp32 ----
  const float4* h4 = (const float4*)(hist + (size_t)b * NL * ND);
  for (int i = tid; i < NL * ND / 4; i += 256) {
    float4 v = h4[i];
    bf16_t* dst = histb + i * 4;
    dst[0] = (bf16_t)v.x; dst[1] = (bf16_t)v.y;
    dst[2] = (bf16_t)v.z; dst[3] = (bf16_t)v.w;
  }
  for (int i = tid; i < 512; i += 256) qi[i] = q_item_g[(size_t)b * 512 + i];
  if (tid < 256) tt[tid] = t_g[(size_t)b * 256 + tid];
  __syncthreads();

  // ---- phase 1: u = hist @ Wkd^T via WMMA (2 M-tiles per wave) ----
  for (int ti = 0; ti < 2; ++ti) {
    int mt = wave * 2 + ti;
    int lr = mt * 16 + lm;                    // A row (M = lane%16)
    v16bf a[4];
#pragma unroll
    for (int kk = 0; kk < 4; ++kk) {
      int kb = kk * 32 + khalf * 16;          // A layout: contiguous K = 16*(lane/16) + s
      v8bf lo = *(const v8bf*)(histb + lr * ND + kb);
      v8bf hi = *(const v8bf*)(histb + lr * ND + kb + 8);
#pragma unroll
      for (int i = 0; i < 8; ++i) { a[kk][i] = lo[i]; a[kk][8 + i] = hi[i]; }
    }
#pragma unroll
    for (int nt = 0; nt < 4; ++nt) {
      int ng = nt * 16 + lm;                  // B col (N = lane%16)
      v8f c = {};
#pragma unroll
      for (int kk = 0; kk < 4; ++kk) {
        v16bf bf = *(const v16bf*)(Wkd + ng * ND + kk * 32 + khalf * 16);
        c = __builtin_amdgcn_wmma_f32_16x16x32_bf16(false, a[kk], false, bf,
                                                    (short)0, c, false, false);
      }
#pragma unroll
      for (int vg = 0; vg < 8; ++vg)          // D: M = vg + 8*(lane/16)
        u[(mt * 16 + vg + 8 * khalf) * NH1 + ng] = c[vg];
    }
  }
  __syncthreads();

  // ---- phase 2: main pair-MLP. 8 M-tiles per wave over 1024 (g,l) rows ----
  const float b3v = sm_b3[0];
  for (int j = 0; j < 8; ++j) {
    int mt = wave * 8 + j;
    int r = mt * 16 + lm, g = r >> 8, l = r & 255;
    v16bf a[4];
#pragma unroll
    for (int kk = 0; kk < 4; ++kk) {          // A = bf16(q[g] * hist[l])
      int kb = kk * 32 + khalf * 16;
      v8bf lo = *(const v8bf*)(histb + l * ND + kb);
      v8bf hi = *(const v8bf*)(histb + l * ND + kb + 8);
      const float* qr = qi + g * ND;
#pragma unroll
      for (int i = 0; i < 8; ++i) {
        a[kk][i]     = (bf16_t)((float)lo[i] * qr[kb + i]);
        a[kk][8 + i] = (bf16_t)((float)hi[i] * qr[kb + 8 + i]);
      }
    }
#pragma unroll
    for (int nt = 0; nt < 4; ++nt) {          // layer 1: C init = t[g]+u[l]
      int ng = nt * 16 + lm;
      v8f c;
#pragma unroll
      for (int vg = 0; vg < 8; ++vg) {
        int rr = mt * 16 + vg + 8 * khalf;
        c[vg] = tt[(rr >> 8) * NH1 + ng] + u[(rr & 255) * NH1 + ng];
      }
#pragma unroll
      for (int kk = 0; kk < 4; ++kk) {
        v16bf bf = *(const v16bf*)(Wm + ng * ND + kk * 32 + khalf * 16);
        c = __builtin_amdgcn_wmma_f32_16x16x32_bf16(false, a[kk], false, bf,
                                                    (short)0, c, false, false);
      }
#pragma unroll
      for (int vg = 0; vg < 8; ++vg)          // relu -> per-wave LDS staging
        h1s[(wave * 16 + vg + 8 * khalf) * NH1 + ng] = (bf16_t)fmaxf(c[vg], 0.f);
    }

    // ---- layer 2 (WMMA, K=64, N=32) + layer 3 (dot with w3) ----
    float sacc[8];
#pragma unroll
    for (int vg = 0; vg < 8; ++vg) sacc[vg] = 0.f;
#pragma unroll
    for (int nt2 = 0; nt2 < 2; ++nt2) {
      int n2 = nt2 * 16 + lm;
      float bb = sm_b2[n2];
      v8f c2;
#pragma unroll
      for (int vg = 0; vg < 8; ++vg) c2[vg] = bb;   // same n for all vg in D layout
#pragma unroll
      for (int kk2 = 0; kk2 < 2; ++kk2) {
        int kb = kk2 * 32 + khalf * 16;
        const bf16_t* hrow = h1s + (wave * 16 + lm) * NH1;
        v8bf lo = *(const v8bf*)(hrow + kb);
        v8bf hi = *(const v8bf*)(hrow + kb + 8);
        v16bf a2;
#pragma unroll
        for (int i = 0; i < 8; ++i) { a2[i] = lo[i]; a2[8 + i] = hi[i]; }
        v16bf bf2 = *(const v16bf*)(W2b + n2 * NH1 + kk2 * 32 + khalf * 16);
        c2 = __builtin_amdgcn_wmma_f32_16x16x32_bf16(false, a2, false, bf2,
                                                     (short)0, c2, false, false);
      }
      float w3v = sm_w3[n2];
#pragma unroll
      for (int vg = 0; vg < 8; ++vg) {        // reduce over 16 N-lanes (half-wave)
        float p = c2[vg] * w3v;
        p += __shfl_xor(p, 1, 32);
        p += __shfl_xor(p, 2, 32);
        p += __shfl_xor(p, 4, 32);
        p += __shfl_xor(p, 8, 32);
        sacc[vg] += p;
      }
    }
    if (lm == 0) {
#pragma unroll
      for (int vg = 0; vg < 8; ++vg) {
        int rr = mt * 16 + vg + 8 * khalf;
        sc[(rr >> 8) * NL + (rr & 255)] = sacc[vg] + b3v;
      }
    }
  }
  __syncthreads();

  // ---- phase 3: masked softmax per (g); waves 0..3 handle g = wave ----
  if (wave < 4) {
    int g = wave;
    const int* mrow = mask + (size_t)b * NL;
    float mx = -1e30f;
    int   mreg[8];
    float sreg[8];
#pragma unroll
    for (int it = 0; it < 8; ++it) {
      int l = lane + 32 * it;
      mreg[it] = mrow[l];
      sreg[it] = sc[g * NL + l];
      if (mreg[it] > 0) mx = fmaxf(mx, sreg[it]);
    }
#pragma unroll
    for (int off = 1; off < 32; off <<= 1) mx = fmaxf(mx, __shfl_xor(mx, off, 32));
    float ereg[8];
    float sum = 0.f;
#pragma unroll
    for (int it = 0; it < 8; ++it) {
      float e = (mreg[it] > 0) ? __expf(sreg[it] - mx) : 0.f;
      ereg[it] = e; sum += e;
    }
#pragma unroll
    for (int off = 1; off < 32; off <<= 1) sum += __shfl_xor(sum, off, 32);
    float inv = (sum > 0.f) ? 1.f / sum : 0.f;  // all-masked row -> zeros
#pragma unroll
    for (int it = 0; it < 8; ++it) sc[g * NL + lane + 32 * it] = ereg[it] * inv;
  }
  __syncthreads();

  // ---- phase 4: weighted pooling (fp32 hist from global/L2), mean over g ----
  {
    int dd = tid & 127, h = tid >> 7;
    const float* hb = hist + (size_t)b * NL * ND;
    float z0 = 0.f, z1 = 0.f;
    for (int l = 0; l < NL; ++l) {
      float v = hb[l * ND + dd];
      z0 += sc[h * NL + l] * v;
      z1 += sc[(h + 2) * NL + l] * v;
    }
    zb[h * ND + dd] = z0;
    zb[(h + 2) * ND + dd] = z1;
  }
  __syncthreads();
  if (tid < 128)
    out[(size_t)b * ND + tid] =
        0.25f * (zb[tid] + zb[128 + tid] + zb[256 + tid] + zb[384 + tid]);
}

// ---------------------------------------------------------------------------
extern "C" void kernel_launch(void* const* d_in, const int* in_sizes, int n_in,
                              void* d_out, int out_size, void* d_ws, size_t ws_size,
                              hipStream_t stream) {
  const float* cand  = (const float*)d_in[0];
  const float* ctx   = (const float*)d_in[1];
  const float* hist  = (const float*)d_in[2];
  const int*   mask  = (const int*)d_in[3];
  const float* qg_w1 = (const float*)d_in[4];
  const float* qg_b1 = (const float*)d_in[5];
  const float* qg_w2 = (const float*)d_in[6];
  const float* qg_b2 = (const float*)d_in[7];
  const float* qg_w3 = (const float*)d_in[8];
  const float* qg_b3 = (const float*)d_in[9];
  const float* q2i_w = (const float*)d_in[10];
  const float* q2i_b = (const float*)d_in[11];
  const float* sm_w1 = (const float*)d_in[12];
  const float* sm_b1 = (const float*)d_in[13];
  const float* sm_w2 = (const float*)d_in[14];
  const float* sm_b2 = (const float*)d_in[15];
  const float* sm_w3 = (const float*)d_in[16];
  const float* sm_b3 = (const float*)d_in[17];

  char* ws = (char*)d_ws;
  bf16_t* Wm       = (bf16_t*)(ws);                    // 16384 B
  bf16_t* Wkd      = (bf16_t*)(ws + 16384);            // 16384 B
  bf16_t* W2b      = (bf16_t*)(ws + 32768);            // 4096  B
  float*  q_item_g = (float*)(ws + 36864);             // 512*4*128*4 = 1 MiB
  float*  t_g      = (float*)(ws + 36864 + 1048576);   // 512*4*64*4  = 512 KiB

  prep_weights<<<1, 256, 0, stream>>>(sm_w1, sm_w2, Wm, Wkd, W2b);
  gen_queries<<<NB, 64, 0, stream>>>(cand, ctx, qg_w1, qg_b1, qg_w2, qg_b2,
                                     qg_w3, qg_b3, q2i_w, q2i_b, sm_w1, sm_b1,
                                     q_item_g, t_g);
  din_main<<<NB, 256, 156672, stream>>>(hist, mask, q_item_g, t_g, Wm, Wkd, W2b,
                                        sm_b2, sm_w3, sm_b3, (float*)d_out);
}
